// DeformableTransformerEncoderLayer_2860448219662
// MI455X (gfx1250) — matrix operand order repaired and checked by reading, not compiled, not run on hardware
//
#include <hip/hip_runtime.h>
#include <hip/hip_bf16.h>

typedef __attribute__((ext_vector_type(16))) _Float16 v16h;
typedef __attribute__((ext_vector_type(8)))  float    v8f;

// ---------------- problem constants ----------------
#define NCAM   6
#define NQ     10000
#define DIM    256
#define HH     58
#define WW     100
#define HWTOK  (HH * WW)          // 5800
#define MV     (NCAM * HWTOK)     // 34800 value tokens
#define MQ     (NCAM * NQ)        // 60000 query rows
#define HEADS  8
#define DH     32                 // == wave32 width
#define P2     8
#define NOFF   (HEADS * P2 * 2)   // 128
#define NATT   (HEADS * P2)       // 64

// ---------------- generic f32 -> f16 cast ----------------
__global__ void k_cast_f16(const float* __restrict__ src, _Float16* __restrict__ dst, int n) {
    int i = blockIdx.x * blockDim.x + threadIdx.x;
    if (i < n) dst[i] = (_Float16)src[i];
}

// ---------------- build value-token activations (f16) ----------------
// X[t, k] = feat0[bn, k, s] + lvl_emb[k] + cam_emb[bn, k], t = bn*HW + s
__global__ void k_build_xval(const float* __restrict__ feat, const float* __restrict__ lvl,
                             const float* __restrict__ cam, _Float16* __restrict__ X) {
    size_t idx = (size_t)blockIdx.x * blockDim.x + threadIdx.x;
    if (idx >= (size_t)MV * DIM) return;
    int k  = (int)(idx & (DIM - 1));
    int t  = (int)(idx >> 8);
    int bn = t / HWTOK;
    int s  = t - bn * HWTOK;
    float v = feat[((size_t)bn * DIM + k) * HWTOK + s] + lvl[k] + cam[bn * DIM + k];
    X[idx] = (_Float16)v;
}

// ---------------- build qp = queries + pos_emb (f16) ----------------
__global__ void k_build_xqp(const float* __restrict__ q, const float* __restrict__ p,
                            _Float16* __restrict__ X) {
    size_t idx = (size_t)blockIdx.x * blockDim.x + threadIdx.x;
    if (idx >= (size_t)MQ * DIM) return;
    X[idx] = (_Float16)(q[idx] + p[idx]);
}

// ---------------- WMMA GEMM: out[M,N] = X[M,K](f16) @ W[K,N](f16) + bias ----------------
// one wave per block; wave computes a 64x32 C tile: 4 A-frags x 2 B-frags,
// 8 v_wmma per K-step. Register budget ~130 VGPRs -> no accumulator spills.
// Row indices are clamped once (no divergence in the K-loop); out-of-range
// rows compute into dead accumulators and are dropped by the guarded
// epilogue. Requires N % 32 == 0 (true for 256/128/64).
__global__ void __launch_bounds__(32, 1)
k_gemm_wmma(const _Float16* __restrict__ X, const _Float16* __restrict__ Wh,
            const float* __restrict__ bias, float* __restrict__ out,
            int M, int N, int K) {
    const int lane  = threadIdx.x;        // 0..31
    const int lhalf = lane >> 4;          // 0/1
    const int lm    = lane & 15;
    const int m0    = blockIdx.x * 64;
    const int n0    = blockIdx.y * 32;

    // clamped row per M-subtile (loads always in-bounds, loop-invariant)
    int mrow[4];
#pragma unroll
    for (int s = 0; s < 4; ++s) {
        int m = m0 + s * 16 + lm;
        mrow[s] = (m < M) ? m : (M - 1);
    }

    v8f acc[4][2] = {};

    for (int k0 = 0; k0 < K; k0 += 32) {
        // A fragments (16x32 f16): lane holds row m, K = lhalf*8 + {0..7, 16..23}
        v16h a[4];
#pragma unroll
        for (int s = 0; s < 4; ++s) {
            const _Float16* xr = X + (size_t)mrow[s] * K + k0 + lhalf * 8;
#pragma unroll
            for (int j = 0; j < 8; ++j) { a[s][j] = xr[j]; a[s][8 + j] = xr[16 + j]; }
        }
        // B fragments (32x16 f16): lane holds col n, K = lhalf*8 + {0..7, 16..23}
        v16h b[2];
#pragma unroll
        for (int t = 0; t < 2; ++t) {
            const _Float16* wr = Wh + (size_t)(k0 + lhalf * 8) * N + n0 + t * 16 + lm;
#pragma unroll
            for (int j = 0; j < 8; ++j) { b[t][j] = wr[(size_t)j * N]; b[t][8 + j] = wr[(size_t)(16 + j) * N]; }
        }
#pragma unroll
        for (int s = 0; s < 4; ++s)
#pragma unroll
            for (int t = 0; t < 2; ++t)
                acc[s][t] = __builtin_amdgcn_wmma_f32_16x16x32_f16(
                    false, a[s], false, b[t], (short)0, acc[s][t], false, false);
    }

    // epilogue: C/D layout — lane: n = n0 + t*16 + lm, rows v + 8*lhalf
    float bv[2];
#pragma unroll
    for (int t = 0; t < 2; ++t) bv[t] = bias[n0 + t * 16 + lm];

#pragma unroll
    for (int s = 0; s < 4; ++s) {
#pragma unroll
        for (int v = 0; v < 8; ++v) {
            int m = m0 + s * 16 + v + 8 * lhalf;
            if (m < M) {
                float* orow = out + (size_t)m * N + n0 + lm;
#pragma unroll
                for (int t = 0; t < 2; ++t) orow[t * 16] = acc[s][t][v] + bv[t];
            }
        }
    }
}

// ---------------- softmax over the 8 sampling points ----------------
__global__ void k_softmax8(float* __restrict__ attn, int ngroups) {
    int g = blockIdx.x * blockDim.x + threadIdx.x;
    if (g >= ngroups) return;
    float* a = attn + (size_t)g * 8;
    float m = a[0];
#pragma unroll
    for (int i = 1; i < 8; ++i) m = fmaxf(m, a[i]);
    float e[8], sum = 0.0f;
#pragma unroll
    for (int i = 0; i < 8; ++i) { e[i] = __expf(a[i] - m); sum += e[i]; }
    float inv = 1.0f / sum;
#pragma unroll
    for (int i = 0; i < 8; ++i) a[i] = e[i] * inv;
}

// ---------------- per-(camera,query) scale = visible / count ----------------
__global__ void k_scale(const int* __restrict__ mask, float* __restrict__ scale) {
    int q = blockIdx.x * blockDim.x + threadIdx.x;
    if (q >= NQ) return;
    int vis[NCAM], cnt = 0;
#pragma unroll
    for (int n = 0; n < NCAM; ++n) {
        int s = 0;
#pragma unroll
        for (int dp = 0; dp < 4; ++dp)
            s += mask[(((size_t)n * NQ + q) * 4 + dp) * 2];
        vis[n] = (s > 0);
        cnt += vis[n];
    }
    float inv = 1.0f / (float)max(cnt, 1);
    for (int n = 0; n < NCAM; ++n) scale[(size_t)n * NQ + q] = vis[n] ? inv : 0.0f;
}

__global__ void k_zero(float* __restrict__ p, int n) {
    int i = blockIdx.x * blockDim.x + threadIdx.x;
    if (i < n) p[i] = 0.0f;
}

// ---------------- deformable sampling + camera aggregation ----------------
// block = one (cam, query) row; wave (32 lanes) = one head; lane = channel-in-head.
__global__ void k_sample(const float* __restrict__ val, const float* __restrict__ off,
                         const float* __restrict__ attn, const float* __restrict__ ref,
                         const float* __restrict__ scale, float* __restrict__ slots) {
    const int r    = blockIdx.x;           // n*NQ + q
    const int h    = threadIdx.x >> 5;
    const int lane = threadIdx.x & 31;
    const int n    = r / NQ;
    const int q    = r - n * NQ;

    const float* offp  = off  + (size_t)r * NOFF + h * 16;
    const float* attp  = attn + (size_t)r * NATT + h * 8;
    const float* refp  = ref  + (size_t)r * 8;
    const float* vbase = val  + (size_t)n * HWTOK * DIM + h * DH + lane;

    float acc = 0.0f;
#pragma unroll
    for (int p = 0; p < P2; ++p) {
        int dp = p & 3;
        // loc*{W,H} - 0.5 simplifies: offsets were divided by norm then re-multiplied
        float x = refp[dp * 2 + 0] * (float)WW + offp[p * 2 + 0] - 0.5f;
        float y = refp[dp * 2 + 1] * (float)HH + offp[p * 2 + 1] - 0.5f;
        float x0f = floorf(x), y0f = floorf(y);
        float fx = x - x0f, fy = y - y0f;
        int x0 = (int)x0f, y0 = (int)y0f;
        float wa = attp[p];
        float w[4] = {(1.f - fx) * (1.f - fy) * wa, fx * (1.f - fy) * wa,
                      (1.f - fx) * fy * wa,         fx * fy * wa};
#pragma unroll
        for (int c = 0; c < 4; ++c) {
            int xi = x0 + (c & 1), yi = y0 + (c >> 1);
            bool valid = (xi >= 0) & (xi < WW) & (yi >= 0) & (yi < HH);
            int xc = min(max(xi, 0), WW - 1);
            int yc = min(max(yi, 0), HH - 1);
            int tok = yc * WW + xc;
            float g = vbase[(size_t)tok * DIM];   // coalesced 128B per wave
            acc += g * (valid ? w[c] : 0.0f);
        }
    }
    float s = scale[r];
    if (s != 0.0f)
        unsafeAtomicAdd(&slots[(size_t)q * DIM + h * DH + lane], acc * s);
}

// ---------------- host side ----------------
static inline size_t align256(size_t x) { return (x + 255) & ~(size_t)255; }

extern "C" void kernel_launch(void* const* d_in, const int* in_sizes, int n_in,
                              void* d_out, int out_size, void* d_ws, size_t ws_size,
                              hipStream_t stream) {
    const float* queries = (const float*)d_in[0];
    const float* pos_emb = (const float*)d_in[1];
    const float* lvl_emb = (const float*)d_in[2];
    const float* cam_emb = (const float*)d_in[3];
    const float* feat0   = (const float*)d_in[4];
    const float* ref3d   = (const float*)d_in[5];
    const int*   bevmask = (const int*)  d_in[6];
    const float* W_value = (const float*)d_in[7];
    const float* b_value = (const float*)d_in[8];
    const float* W_off   = (const float*)d_in[9];
    const float* b_off   = (const float*)d_in[10];
    const float* W_attn  = (const float*)d_in[11];
    const float* b_attn  = (const float*)d_in[12];
    const float* W_out   = (const float*)d_in[13];
    const float* b_out   = (const float*)d_in[14];
    float* out = (float*)d_out;

    char* ws = (char*)d_ws;
    size_t o = 0;
    _Float16* whv   = (_Float16*)(ws + o); o = align256(o + (size_t)DIM * DIM  * 2);
    _Float16* who   = (_Float16*)(ws + o); o = align256(o + (size_t)DIM * NOFF * 2);
    _Float16* wha   = (_Float16*)(ws + o); o = align256(o + (size_t)DIM * NATT * 2);
    _Float16* whw   = (_Float16*)(ws + o); o = align256(o + (size_t)DIM * DIM  * 2);
    _Float16* xval  = (_Float16*)(ws + o); o = align256(o + (size_t)MV * DIM * 2);
    _Float16* xqp   = (_Float16*)(ws + o); o = align256(o + (size_t)MQ * DIM * 2);
    float*    val   = (float*)   (ws + o); o = align256(o + (size_t)MV * DIM * 4);
    float*    offb  = (float*)   (ws + o); o = align256(o + (size_t)MQ * NOFF * 4);
    float*    attnb = (float*)   (ws + o); o = align256(o + (size_t)MQ * NATT * 4);
    float*    scale = (float*)   (ws + o); o = align256(o + (size_t)MQ * 4);
    float*    slots = (float*)   (ws + o); o = align256(o + (size_t)NQ * DIM * 4);
    _Float16* xslot = (_Float16*)(ws + o); o = align256(o + (size_t)NQ * DIM * 2);
    (void)ws_size;

    const int T = 256;

    // 1) weights -> f16
    k_cast_f16<<<(DIM * DIM  + T - 1) / T, T, 0, stream>>>(W_value, whv, DIM * DIM);
    k_cast_f16<<<(DIM * NOFF + T - 1) / T, T, 0, stream>>>(W_off,   who, DIM * NOFF);
    k_cast_f16<<<(DIM * NATT + T - 1) / T, T, 0, stream>>>(W_attn,  wha, DIM * NATT);
    k_cast_f16<<<(DIM * DIM  + T - 1) / T, T, 0, stream>>>(W_out,   whw, DIM * DIM);

    // 2) activations -> f16
    k_build_xval<<<(int)(((size_t)MV * DIM + T - 1) / T), T, 0, stream>>>(feat0, lvl_emb, cam_emb, xval);
    k_build_xqp <<<(int)(((size_t)MQ * DIM + T - 1) / T), T, 0, stream>>>(queries, pos_emb, xqp);

    // 3) WMMA GEMMs: value / offsets / attention logits (64x32 tile per wave)
    {
        dim3 g((MV + 63) / 64, DIM / 32);
        k_gemm_wmma<<<g, 32, 0, stream>>>(xval, whv, b_value, val, MV, DIM, DIM);
    }
    {
        dim3 g((MQ + 63) / 64, NOFF / 32);
        k_gemm_wmma<<<g, 32, 0, stream>>>(xqp, who, b_off, offb, MQ, NOFF, DIM);
    }
    {
        dim3 g((MQ + 63) / 64, NATT / 32);
        k_gemm_wmma<<<g, 32, 0, stream>>>(xqp, wha, b_attn, attnb, MQ, NATT, DIM);
    }

    // 4) softmax over the 8 points per (row, head)
    k_softmax8<<<(MQ * HEADS + T - 1) / T, T, 0, stream>>>(attnb, MQ * HEADS);

    // 5) visibility scale + zero accumulation buffer
    k_scale<<<(NQ + T - 1) / T, T, 0, stream>>>(bevmask, scale);
    k_zero <<<(NQ * DIM + T - 1) / T, T, 0, stream>>>(slots, NQ * DIM);

    // 6) bilinear deformable sampling, camera-aggregated via f32 atomics
    k_sample<<<MQ, HEADS * 32, 0, stream>>>(val, offb, attnb, ref3d, scale, slots);

    // 7) output projection (WMMA)
    k_cast_f16<<<(NQ * DIM + T - 1) / T, T, 0, stream>>>(slots, xslot, NQ * DIM);
    {
        dim3 g((NQ + 63) / 64, DIM / 32);
        k_gemm_wmma<<<g, 32, 0, stream>>>(xslot, whw, b_out, out, NQ, DIM, DIM);
    }
}
